// SelfAttention_83992380441159
// MI455X (gfx1250) — matrix operand order repaired and checked by reading, not compiled, not run on hardware
//
#include <hip/hip_runtime.h>
#include <hip/hip_bf16.h>

typedef _Float16 h16 __attribute__((ext_vector_type(16)));
typedef _Float16 h8  __attribute__((ext_vector_type(8)));
typedef float    f8  __attribute__((ext_vector_type(8)));

#define BATCH 4
#define SEQ   2048
#define DIN   1024
#define EMB   64
#define NSPLIT 4
#define KSPAN (SEQ / NSPLIT)  // 512 keys per split
#define NJOBS (BATCH * (SEQ / 16) * NSPLIT)  // 2048 attention waves

static __device__ __forceinline__ f8 wmma_f16(h16 a, h16 b, f8 c) {
  // D = A(16x32 f16) * B(32x16 f16) + C(16x16 f32)
  return __builtin_amdgcn_wmma_f32_16x16x32_f16(false, a, false, b, (short)0, c,
                                                false, false);
}

static __device__ __forceinline__ h16 cvt_h16(f8 lo, f8 hi) {
  h16 r;
#pragma unroll
  for (int i = 0; i < 8; ++i) {
    r[i]     = (_Float16)lo[i];
    r[i + 8] = (_Float16)hi[i];
  }
  return r;
}

// ---------------------------------------------------------------------------
// Kernel 1: QKV projection, software-pipelined.  One wave per
// (b, 16-row seq tile, 16-col n tile); raw f32 vectors for chunk k+32 are
// loaded while chunk k is converted and fed to 3 WMMAs.
// Q,K stored f16 row-major [b][s][64]; V stored f16 transposed [b][64][s].
// ---------------------------------------------------------------------------
static __device__ __forceinline__ void load_raw(const float* xr,
                                                const float* wqr,
                                                const float* wkr,
                                                const float* wvr, int k0,
                                                int half, f8* r) {
  r[0] = *(const f8*)(xr + k0 + (half << 4));         // A halves 0-7
  r[1] = *(const f8*)(xr + k0 + (half << 4) + 8);     // A halves 8-15
  r[2] = *(const f8*)(wqr + k0 + (half << 4));        // B halves 0-7
  r[3] = *(const f8*)(wqr + k0 + (half << 4) + 8);    // B halves 8-15
  r[4] = *(const f8*)(wkr + k0 + (half << 4));
  r[5] = *(const f8*)(wkr + k0 + (half << 4) + 8);
  r[6] = *(const f8*)(wvr + k0 + (half << 4));
  r[7] = *(const f8*)(wvr + k0 + (half << 4) + 8);
}

__global__ __launch_bounds__(256) void qkv_proj_kernel(
    const float* __restrict__ x, const float* __restrict__ Wq,
    const float* __restrict__ Wk, const float* __restrict__ Wv,
    _Float16* __restrict__ Qh, _Float16* __restrict__ Kh,
    _Float16* __restrict__ Vt) {
  const int wave = (int)((blockIdx.x * blockDim.x + threadIdx.x) >> 5);
  const int lane = (int)(threadIdx.x & 31);
  const int ntile = wave & 3;
  const int srow  = (wave >> 2) & 127;
  const int b     = wave >> 9;
  const int s0 = srow << 4, n0 = ntile << 4;
  const int l16 = lane & 15, half = lane >> 4;

  const float* xr  = x  + ((size_t)b * SEQ + s0 + l16) * DIN;
  const float* wqr = Wq + (size_t)(n0 + l16) * DIN;
  const float* wkr = Wk + (size_t)(n0 + l16) * DIN;
  const float* wvr = Wv + (size_t)(n0 + l16) * DIN;

  f8 aq = {}, ak = {}, av = {};
  f8 bufA[8], bufB[8];
  load_raw(xr, wqr, wkr, wvr, 0, half, bufA);

  for (int k0 = 0; k0 < DIN; k0 += 64) {
    load_raw(xr, wqr, wkr, wvr, k0 + 32, half, bufB);
    {
      h16 a  = cvt_h16(bufA[0], bufA[1]);
      h16 bq = cvt_h16(bufA[2], bufA[3]);
      h16 bk = cvt_h16(bufA[4], bufA[5]);
      h16 bv = cvt_h16(bufA[6], bufA[7]);
      aq = wmma_f16(a, bq, aq);
      ak = wmma_f16(a, bk, ak);
      av = wmma_f16(a, bv, av);
    }
    if (k0 + 64 < DIN) load_raw(xr, wqr, wkr, wvr, k0 + 64, half, bufA);
    {
      h16 a  = cvt_h16(bufB[0], bufB[1]);
      h16 bq = cvt_h16(bufB[2], bufB[3]);
      h16 bk = cvt_h16(bufB[4], bufB[5]);
      h16 bv = cvt_h16(bufB[6], bufB[7]);
      aq = wmma_f16(a, bq, aq);
      ak = wmma_f16(a, bk, ak);
      av = wmma_f16(a, bv, av);
    }
  }

  // C/D layout: row = r + 8*half, col = l16.
#pragma unroll
  for (int r = 0; r < 8; ++r) {
    const size_t idx =
        ((size_t)b * SEQ + s0 + (half << 3) + r) * EMB + n0 + l16;
    Qh[idx] = (_Float16)aq[r];
    Kh[idx] = (_Float16)ak[r];
  }
  h8 vp;
#pragma unroll
  for (int r = 0; r < 8; ++r) vp[r] = (_Float16)av[r];
  *(h8*)(Vt + ((size_t)b * EMB + n0 + l16) * SEQ + s0 + (half << 3)) = vp;
}

// ---------------------------------------------------------------------------
// Kernel 2: flash attention with 4-way key split (2048 waves).  Each wave
// handles one (b, q-tile, key-split); K B-fragments are double-buffered and V
// B-fragments are loaded before the softmax VALU work.  Stores unnormalized
// partial O' plus per-row (m, l) for the combine kernel.
// ---------------------------------------------------------------------------
__global__ __launch_bounds__(256) void flash_attn_kernel(
    const _Float16* __restrict__ Qh, const _Float16* __restrict__ Kh,
    const _Float16* __restrict__ Vt, float* __restrict__ pacc,
    float* __restrict__ pm, float* __restrict__ pl) {
  __shared__ _Float16 pstage[8][16 * 32];  // per-wave 16x32 P staging
  const int wave = (int)((blockIdx.x * blockDim.x + threadIdx.x) >> 5);
  const int wl   = (int)(threadIdx.x >> 5) & 7;
  const int lane = (int)(threadIdx.x & 31);
  const int split = wave & 3;
  const int qt    = (wave >> 2) & 127;
  const int b     = wave >> 9;
  const int s0 = qt << 4;
  const int l16 = lane & 15, half = lane >> 4;
  const int kbeg = split * KSPAN, kend = kbeg + KSPAN;

  const _Float16* kB = Kh + (size_t)b * SEQ * EMB;
  const _Float16* vB = Vt + (size_t)b * EMB * SEQ;

  // Q A-fragments (E=64 -> two K=32 chunks), resident for the whole sweep.
  const _Float16* qr = Qh + ((size_t)b * SEQ + s0 + l16) * EMB;
  h16 qa[2];
#pragma unroll
  for (int c = 0; c < 2; ++c)
#pragma unroll
    for (int h = 0; h < 16; ++h)
      qa[c][h] = qr[(c << 5) + (half << 4) + h];

  f8 acc[4] = {{}, {}, {}, {}};
  float mrow[8], lrow[8];
#pragma unroll
  for (int r = 0; r < 8; ++r) { mrow[r] = -1e30f; lrow[r] = 0.f; }

  _Float16* pw = &pstage[wl][0];

  auto load_kf = [&](int kb, h16 (&kf)[2][2]) {
#pragma unroll
    for (int t = 0; t < 2; ++t)
#pragma unroll
      for (int c = 0; c < 2; ++c)
        kf[t][c] = *(const h16*)(kB + (size_t)(kb + (t << 4) + l16) * EMB +
                                 (c << 5) + (half << 4));
  };

  auto process = [&](int kb, h16 (&kf)[2][2], h16 (&kfn)[2][2], int kbn) {
    // scores for 32 keys with the current (already loaded) K fragments.
    f8 sc[2];
#pragma unroll
    for (int t = 0; t < 2; ++t) {
      f8 st = {};
#pragma unroll
      for (int c = 0; c < 2; ++c) st = wmma_f16(qa[c], kf[t][c], st);
      sc[t] = st;
    }
    // issue next-chunk K loads and this chunk's V loads; the softmax VALU
    // work below overlaps their latency.
    load_kf(kbn, kfn);
    h16 vf[4];
#pragma unroll
    for (int e = 0; e < 4; ++e)
      vf[e] = *(const h16*)(vB + (size_t)((e << 4) + l16) * SEQ + kb +
                            (half << 4));
    __builtin_prefetch((const void*)(vB + (size_t)l16 * SEQ + kb + 32), 0, 0);

    // online softmax; row = r + 8*half, cols striped over 16-lane halves.
#pragma unroll
    for (int r = 0; r < 8; ++r) {
      const float s0v = sc[0][r] * 0.125f;  // 1/sqrt(64)
      const float s1v = sc[1][r] * 0.125f;
      float mx = fmaxf(s0v, s1v);
#pragma unroll
      for (int d = 1; d < 16; d <<= 1) mx = fmaxf(mx, __shfl_xor(mx, d, 32));
      const float mn = fmaxf(mrow[r], mx);
      const float alpha = __expf(mrow[r] - mn);
      const float p0 = __expf(s0v - mn), p1 = __expf(s1v - mn);
      float rs = p0 + p1;
#pragma unroll
      for (int d = 1; d < 16; d <<= 1) rs += __shfl_xor(rs, d, 32);
      lrow[r] = lrow[r] * alpha + rs;
      mrow[r] = mn;
#pragma unroll
      for (int e = 0; e < 4; ++e) acc[e][r] *= alpha;
      const int prow = (half << 3) + r;
      pw[prow * 32 + l16]      = (_Float16)p0;
      pw[prow * 32 + 16 + l16] = (_Float16)p1;
    }
    asm volatile("s_wait_dscnt 0" ::: "memory");  // wave-local LDS RAW fence

    // reload P as an A fragment (16x32 f16) and accumulate P @ V.
    h16 pa;
#pragma unroll
    for (int h = 0; h < 16; ++h)
      pa[h] = pw[l16 * 32 + (half << 4) + h];
#pragma unroll
    for (int e = 0; e < 4; ++e) acc[e] = wmma_f16(pa, vf[e], acc[e]);
  };

  h16 kf0[2][2], kf1[2][2];
  load_kf(kbeg, kf0);
  for (int kb = kbeg; kb < kend; kb += 64) {
    process(kb, kf0, kf1, kb + 32);
    const int kbn = (kb + 64 < kend) ? (kb + 64) : (kend - 32);
    process(kb + 32, kf1, kf0, kbn);
  }

  // store partials: O' (C/D layout), and (m, l) once per row.
  const size_t jrow = (size_t)wave * 16;
#pragma unroll
  for (int r = 0; r < 8; ++r)
#pragma unroll
    for (int e = 0; e < 4; ++e)
      pacc[(jrow + (half << 3) + r) * EMB + (e << 4) + l16] = acc[e][r];
  if (l16 == 0) {
#pragma unroll
    for (int r = 0; r < 8; ++r) {
      pm[jrow + (half << 3) + r] = mrow[r];
      pl[jrow + (half << 3) + r] = lrow[r];
    }
  }
}

// ---------------------------------------------------------------------------
// Kernel 3: combine the 4 key-split partials.  One thread per output element.
// ---------------------------------------------------------------------------
__global__ __launch_bounds__(256) void combine_kernel(
    const float* __restrict__ pacc, const float* __restrict__ pm,
    const float* __restrict__ pl, float* __restrict__ out) {
  const int idx = (int)(blockIdx.x * blockDim.x + threadIdx.x);
  const int e   = idx & 63;
  const int gs  = idx >> 6;          // b*SEQ + s
  const int row = gs & 15;
  const int qt  = (gs >> 4) & 127;
  const int b   = gs >> 11;
  const int jbase = ((b << 7) | qt) << 2;  // first of 4 split jobs

  float mv[NSPLIT];
  float m = -1e30f;
#pragma unroll
  for (int i = 0; i < NSPLIT; ++i) {
    mv[i] = pm[(jbase + i) * 16 + row];
    m = fmaxf(m, mv[i]);
  }
  float num = 0.f, den = 0.f;
#pragma unroll
  for (int i = 0; i < NSPLIT; ++i) {
    const float w = __expf(mv[i] - m);
    num += w * pacc[((size_t)(jbase + i) * 16 + row) * EMB + e];
    den += w * pl[(jbase + i) * 16 + row];
  }
  out[idx] = num / den;
}

extern "C" void kernel_launch(void* const* d_in, const int* in_sizes, int n_in,
                              void* d_out, int out_size, void* d_ws,
                              size_t ws_size, hipStream_t stream) {
  (void)in_sizes; (void)n_in; (void)out_size; (void)ws_size;
  const float* x  = (const float*)d_in[0];
  const float* Wq = (const float*)d_in[1];
  const float* Wk = (const float*)d_in[2];
  const float* Wv = (const float*)d_in[3];
  float* out = (float*)d_out;

  const size_t qkv_elems = (size_t)BATCH * SEQ * EMB;  // 524288
  _Float16* Qh = (_Float16*)d_ws;
  _Float16* Kh = Qh + qkv_elems;
  _Float16* Vt = Kh + qkv_elems;
  float* pacc = (float*)(Vt + qkv_elems);          // 2048*16*64 f32 = 8 MB
  float* pm   = pacc + (size_t)NJOBS * 16 * EMB;   // 2048*16 f32
  float* pl   = pm + (size_t)NJOBS * 16;           // 2048*16 f32

  // 2048 waves (one per QKV tile triple) -> 256 blocks of 8 waves.
  qkv_proj_kernel<<<256, 256, 0, stream>>>(x, Wq, Wk, Wv, Qh, Kh, Vt);
  // 2048 waves (one per q-tile x key-split) -> 256 blocks of 8 waves.
  flash_attn_kernel<<<256, 256, 0, stream>>>(Qh, Kh, Vt, pacc, pm, pl);
  // 524288 output elements -> 2048 blocks of 256 threads.
  combine_kernel<<<2048, 256, 0, stream>>>(pacc, pm, pl, out);
}
